// FC_MT_LSTM_37125697307351
// MI455X (gfx1250) — matrix operand order repaired and checked by reading, not compiled, not run on hardware
//
#include <hip/hip_runtime.h>

// ---------------------------------------------------------------------------
// FC_MT_LSTM for MI455X (gfx1250).
// bf16 WMMA everywhere; operands pre-converted to bf16 so fragment feeds are
// pure b128 loads; LSTM recurrence keeps weights in VGPRs, h/c on-chip, and
// double-buffers the xg tile in LDS via global_load_async_to_lds_b128.
// ---------------------------------------------------------------------------

typedef __attribute__((ext_vector_type(16))) __bf16          v16bf;
typedef __attribute__((ext_vector_type(8)))  float           v8f;
typedef __attribute__((ext_vector_type(4)))  unsigned int    u32x4;
typedef __attribute__((ext_vector_type(4)))  float           f32x4;
typedef __attribute__((ext_vector_type(4)))  unsigned short  u16x4;

constexpr int B_  = 128;            // batch
constexpr int T_  = 512;            // time
constexpr int BT_ = B_ * T_;        // 65536
constexpr int H_  = 128;            // hidden
constexpr int G_  = 4 * H_;         // 512 gates

// ---- bf16 helpers ----------------------------------------------------------
static __device__ __forceinline__ unsigned short f2bfu(float f) {
  unsigned u = __builtin_bit_cast(unsigned, f);
  u += 0x7FFFu + ((u >> 16) & 1u);
  return (unsigned short)(u >> 16);
}
static __device__ __forceinline__ float bfu2f(unsigned short s) {
  unsigned u = ((unsigned)s) << 16;
  return __builtin_bit_cast(float, u);
}
static __device__ __forceinline__ float sigf(float x) { return 1.0f / (1.0f + expf(-x)); }

struct B32x8 { u32x4 lo, hi; };
static __device__ __forceinline__ v16bf mk16(u32x4 lo, u32x4 hi) {
  B32x8 t{lo, hi};
  return __builtin_bit_cast(v16bf, t);
}

// A fragment (16x32 bf16): lane m=lane&15, khi=(lane>>4)*8;
// elements 0..7 = K[k0..k0+7], elements 8..15 = K[k0+16..k0+23]
static __device__ __forceinline__ v16bf load_a_bf16(const unsigned short* row, int k0) {
  return mk16(*(const u32x4*)(row + k0), *(const u32x4*)(row + k0 + 16));
}
// B fragment (32x16 bf16): lane n=lane&15, k0 += (lane>>4)*8;
// elements 0..7 = K[k0..k0+7], elements 8..15 = K[k0+16..k0+23]
static __device__ __forceinline__ v16bf load_b_bf16(const unsigned short* row, int k0) {
  return mk16(*(const u32x4*)(row + k0), *(const u32x4*)(row + k0 + 16));
}

// LDS byte address = low 32 bits of the generic pointer (flat aperture).
static __device__ __forceinline__ unsigned lds_addr_of(const void* p) {
  return (unsigned)(uintptr_t)p;
}
static __device__ __forceinline__ void async_b128(unsigned lds, const void* g) {
  asm volatile("global_load_async_to_lds_b128 %0, %1, off"
               :: "v"(lds), "v"((unsigned long long)(uintptr_t)g) : "memory");
}

// ===========================================================================
// Pre-passes: f32 -> bf16 conversions
// ===========================================================================
__global__ void __launch_bounds__(256)
f32_to_bf16(const float* __restrict__ in, unsigned short* __restrict__ out) {
  long i = ((long)blockIdx.x * 256 + threadIdx.x) * 4;
  f32x4 v = *(const f32x4*)(in + i);
  u16x4 o;
#pragma unroll
  for (int k = 0; k < 4; ++k) o[k] = f2bfu(v[k]);
  *(u16x4*)(out + i) = o;
}

// conv weight (O, I, 3) f32 -> (3, O, I) bf16
__global__ void __launch_bounds__(256)
convw_tr_bf16(const float* __restrict__ w, unsigned short* __restrict__ wt,
              int O, int I) {
  int idx = blockIdx.x * 256 + threadIdx.x;
  int total = O * I * 3;
  if (idx >= total) return;
  int tap = idx % 3;
  int rest = idx / 3;
  int i = rest % I;
  int o = rest / I;
  wt[((size_t)tap * O + o) * I + i] = f2bfu(w[idx]);
}

// ===========================================================================
// Conv1d(k=3, pad=1) as 3-tap accumulated WMMA GEMM (bf16 operands).
// Xb: (B*T, KIN) bf16. Wt: (3, NOUT, KIN) bf16. Y: (B*T, NOUT) f32.
// ===========================================================================
template <int KIN, int NOUT>
__global__ void __launch_bounds__(32)
conv3_wmma(const unsigned short* __restrict__ Xb, const unsigned short* __restrict__ Wt,
           const float* __restrict__ bias, float* __restrict__ Y) {
  const int lane = threadIdx.x;
  const int m0   = blockIdx.x * 16;
  constexpr int JT = NOUT / 16;
  v8f acc[JT] = {};

  const int mrow = lane & 15;
  const int khiA = (lane >> 4) << 3;
  const int kbB  = (lane >> 4) << 3;
  const int g    = m0 + mrow;
  const int t    = g & (T_ - 1);
  const u32x4 zz = {0u, 0u, 0u, 0u};

#pragma unroll
  for (int dtap = 0; dtap < 3; ++dtap) {
    const int dt = dtap - 1;
    const bool ok = ((unsigned)(t + dt)) < (unsigned)T_;
    const unsigned short* row = Xb + (size_t)(g + dt) * KIN;
#pragma unroll
    for (int ks = 0; ks < KIN / 32; ++ks) {
      int k0 = 32 * ks + khiA;
      u32x4 lo = ok ? *(const u32x4*)(row + k0)      : zz;
      u32x4 hi = ok ? *(const u32x4*)(row + k0 + 16) : zz;
      v16bf a = mk16(lo, hi);
#pragma unroll
      for (int j = 0; j < JT; ++j) {
        const unsigned short* brow =
            Wt + ((size_t)dtap * NOUT + 16 * j + (lane & 15)) * KIN;
        v16bf bm = load_b_bf16(brow, 32 * ks + kbB);
        acc[j] = __builtin_amdgcn_wmma_f32_16x16x32_bf16(
            false, a, false, bm, (short)0, acc[j], false, false);
      }
    }
  }
  const int colD = lane & 15;
  const int rb   = (lane >> 4) << 3;
#pragma unroll
  for (int j = 0; j < JT; ++j) {
    int n = 16 * j + colD;
    float bs = bias[n];
#pragma unroll
    for (int r = 0; r < 8; ++r)
      Y[(size_t)(m0 + rb + r) * NOUT + n] = acc[j][r] + bs;
  }
}

// ===========================================================================
// BatchNorm stats + fused normalize/ReLU (writes bf16 activation)
// ===========================================================================
__global__ void __launch_bounds__(256)
bn_stats(const float* __restrict__ Y, int C, int n,
         float* __restrict__ mean, float* __restrict__ rstd) {
  const int c = blockIdx.x;
  float s = 0.f, s2 = 0.f;
  for (int i = threadIdx.x; i < n; i += 256) {
    float v = Y[(size_t)i * C + c];
    s += v; s2 += v * v;
  }
  __shared__ float sh[256], sh2[256];
  sh[threadIdx.x] = s; sh2[threadIdx.x] = s2;
  __syncthreads();
  for (int off = 128; off > 0; off >>= 1) {
    if (threadIdx.x < off) {
      sh[threadIdx.x]  += sh[threadIdx.x + off];
      sh2[threadIdx.x] += sh2[threadIdx.x + off];
    }
    __syncthreads();
  }
  if (threadIdx.x == 0) {
    float m   = sh[0] / (float)n;
    float var = sh2[0] / (float)n - m * m;
    mean[c] = m;
    rstd[c] = rsqrtf(var + 1e-5f);
  }
}

__global__ void __launch_bounds__(256)
bn_relu_bf16(const float* __restrict__ Y, int C,
             const float* __restrict__ mean, const float* __restrict__ rstd,
             const float* __restrict__ gam, const float* __restrict__ bet,
             unsigned short* __restrict__ out) {
  long i = (long)blockIdx.x * 256 + threadIdx.x;
  int c = (int)(i % C);
  float v = (Y[i] - mean[c]) * rstd[c] * gam[c] + bet[c];
  out[i] = f2bfu(v > 0.f ? v : 0.f);
}

// ===========================================================================
// Gate GEMM: XG = X @ W^T + bias (bf16 in, bf16 out).
// X:(BT,K) bf16 row-major (rows are b*T+t).  W:(512,K) bf16.
// Output in scan-friendly layout: XG[(t*B + b)*512 + n].
// ===========================================================================
__global__ void __launch_bounds__(32)
gate_gemm(const unsigned short* __restrict__ X, int K,
          const unsigned short* __restrict__ W, const float* __restrict__ bias,
          unsigned short* __restrict__ XG) {
  const int lane = threadIdx.x;
  const int m0   = blockIdx.x * 16;
  const int n0   = blockIdx.y * 64;
  v8f acc[4] = {};

  const int khiA = (lane >> 4) << 3;
  const int kbB  = (lane >> 4) << 3;
  const unsigned short* arow = X + (size_t)(m0 + (lane & 15)) * K;
  const unsigned short* brow[4];
#pragma unroll
  for (int j = 0; j < 4; ++j)
    brow[j] = W + (size_t)(n0 + 16 * j + (lane & 15)) * K;

  for (int ks = 0; ks < K; ks += 32) {
    v16bf a = load_a_bf16(arow, ks + khiA);
#pragma unroll
    for (int j = 0; j < 4; ++j) {
      v16bf bm = load_b_bf16(brow[j], ks + kbB);
      acc[j] = __builtin_amdgcn_wmma_f32_16x16x32_bf16(
          false, a, false, bm, (short)0, acc[j], false, false);
    }
  }
  const int colD = lane & 15;
  const int rb   = (lane >> 4) << 3;
#pragma unroll
  for (int j = 0; j < 4; ++j) {
    int n = n0 + 16 * j + colD;
    float bs = bias[n];
#pragma unroll
    for (int r = 0; r < 8; ++r) {
      int grow = m0 + rb + r;                 // b*T + t
      size_t orow = (size_t)(grow & (T_ - 1)) * B_ + (grow >> 9);
      XG[orow * G_ + n] = f2bfu(acc[j][r] + bs);
    }
  }
}

// ===========================================================================
// LSTM scan. One WG per (16-batch chunk, direction); 8 waves.
// Wave w owns gate tiles n = {w*16, 128+w*16, 256+w*16, 384+w*16} so the
// i/f/g/o quadruple for each hidden unit lands in one wave -> cell update is
// register-local; c lives in VGPRs; h (bf16) in LDS; whh fragments in VGPRs.
// xg tile (16x512 bf16 = 16KB) double-buffered in LDS via async loads.
// XG:(2,T,B,512) bf16.  WHHb:(2,512,128) bf16.
// Hf:(B,T,256) f32 or null.  Hb:(B,T,256) bf16 or null.
// ===========================================================================
__global__ void __launch_bounds__(256)
lstm_scan(const unsigned short* __restrict__ XG, const unsigned short* __restrict__ WHHb,
          float* __restrict__ Hf, unsigned short* __restrict__ Hb) {
  const int tid  = threadIdx.x;
  const int lane = tid & 31;
  const int wave = tid >> 5;
  const int dir  = blockIdx.x >> 3;
  const int b0   = (blockIdx.x & 7) * 16;

  const unsigned short* xg  = XG   + (size_t)dir * T_ * B_ * G_;
  const unsigned short* whh = WHHb + (size_t)dir * G_ * H_;

  __shared__ __align__(16) unsigned short hbuf[16][H_];       //  4 KB
  __shared__ __align__(16) unsigned short xbuf[2][16][G_];    // 32 KB

  for (int i = tid; i < 16 * H_ / 2; i += 256) ((unsigned*)hbuf)[i] = 0u;

  const int colD = lane & 15;
  const int rb   = (lane >> 4) << 3;
  const int mA   = lane & 15;
  const int khiA = (lane >> 4) << 3;
  const int kbB  = (lane >> 4) << 3;
  const int jh   = wave * 16 + colD;          // hidden unit owned by this lane

  // recurrent-weight fragments (constant over time)
  v16bf bw[4][4];
#pragma unroll
  for (int gq = 0; gq < 4; ++gq) {
    const unsigned short* brow = whh + (size_t)(gq * H_ + wave * 16 + (lane & 15)) * H_;
#pragma unroll
    for (int s = 0; s < 4; ++s)
      bw[gq][s] = load_b_bf16(brow, 32 * s + kbB);
  }

  v8f creg = {};
  const unsigned xlds0 = lds_addr_of(&xbuf[0][0][0]);
  const unsigned xlds1 = lds_addr_of(&xbuf[1][0][0]);

  // prologue: async-fetch step-0 tile
  {
    const int t0 = dir ? (T_ - 1) : 0;
    const char* src = (const char*)(xg + ((size_t)t0 * B_ + b0) * G_);
#pragma unroll
    for (int q = 0; q < 4; ++q) {
      unsigned off = (unsigned)tid * 16u + (unsigned)q * 4096u;
      async_b128(xlds0 + off, src + off);
    }
  }

  for (int step = 0; step < T_; ++step) {
    const int t = dir ? (T_ - 1 - step) : step;
    const unsigned short (*xb)[G_] = xbuf[step & 1];

    asm volatile("s_wait_asynccnt 0x0" ::: "memory");
    __syncthreads();                          // xb tile + h_{t-1} visible to all

    if (step + 1 < T_) {                      // prefetch next tile (overlapped)
      const int tn = dir ? (t - 1) : (t + 1);
      const unsigned xnext = (step & 1) ? xlds0 : xlds1;
      const char* src = (const char*)(xg + ((size_t)tn * B_ + b0) * G_);
#pragma unroll
      for (int q = 0; q < 4; ++q) {
        unsigned off = (unsigned)tid * 16u + (unsigned)q * 4096u;
        async_b128(xnext + off, src + off);
      }
    }

    // acc <- xg tile (D layout), gate gq at column gq*128 + jh
    v8f acc[4];
#pragma unroll
    for (int gq = 0; gq < 4; ++gq)
#pragma unroll
      for (int r = 0; r < 8; ++r)
        acc[gq][r] = bfu2f(xb[rb + r][gq * H_ + jh]);

    // gates += h @ whh^T  (K=128, 4 bf16 WMMA steps, 16 WMMAs)
#pragma unroll
    for (int s = 0; s < 4; ++s) {
      int k0 = 32 * s + khiA;
      v16bf a = mk16(*(const u32x4*)(&hbuf[mA][k0]),
                     *(const u32x4*)(&hbuf[mA][k0 + 16]));
#pragma unroll
      for (int gq = 0; gq < 4; ++gq)
        acc[gq] = __builtin_amdgcn_wmma_f32_16x16x32_bf16(
            false, a, false, bw[gq][s], (short)0, acc[gq], false, false);
    }
    __syncthreads();                          // all waves done reading h_{t-1}

    // register-local cell update: lane owns column jh, rows rb..rb+7
#pragma unroll
    for (int r = 0; r < 8; ++r) {
      int m = rb + r;
      float c = sigf(acc[1][r]) * creg[r] + sigf(acc[0][r]) * tanhf(acc[2][r]);
      float h = sigf(acc[3][r]) * tanhf(c);
      creg[r] = c;
      unsigned short hu = f2bfu(h);
      hbuf[m][jh] = hu;
      size_t o = ((size_t)(b0 + m) * T_ + t) * (2 * H_) + (size_t)dir * H_ + jh;
      if (Hf) Hf[o] = h;
      if (Hb) Hb[o] = hu;
    }
  }
}

// ===========================================================================
// Attention: scores, softmax over T, attended vector.
// ===========================================================================
__global__ void __launch_bounds__(256)
att_scores(const float* __restrict__ Hm, const float* __restrict__ w,
           const float* __restrict__ ab, float* __restrict__ scores) {
  long i = (long)blockIdx.x * 256 + threadIdx.x;
  const float* row = Hm + i * (2 * H_);
  float s = 0.f;
  for (int d = 0; d < 2 * H_; ++d) s += row[d] * w[d];
  scores[i] = s + ab[0];
}

__global__ void __launch_bounds__(256)
softmax_attend(const float* __restrict__ scores, const float* __restrict__ Hm,
               float* __restrict__ attn, float* __restrict__ attended) {
  const int b = blockIdx.x, tid = threadIdx.x;
  __shared__ float p[T_];
  __shared__ float red[256];

  float mx = -3.4e38f;
  for (int t = tid; t < T_; t += 256) mx = fmaxf(mx, scores[(long)b * T_ + t]);
  red[tid] = mx; __syncthreads();
  for (int off = 128; off > 0; off >>= 1) {
    if (tid < off) red[tid] = fmaxf(red[tid], red[tid + off]);
    __syncthreads();
  }
  mx = red[0]; __syncthreads();

  float sum = 0.f;
  for (int t = tid; t < T_; t += 256) {
    float e = expf(scores[(long)b * T_ + t] - mx);
    p[t] = e; sum += e;
  }
  red[tid] = sum; __syncthreads();
  for (int off = 128; off > 0; off >>= 1) {
    if (tid < off) red[tid] += red[tid + off];
    __syncthreads();
  }
  float inv = 1.f / red[0];
  __syncthreads();

  for (int t = tid; t < T_; t += 256) {
    float v = p[t] * inv;
    p[t] = v;
    attn[(long)b * T_ + t] = v;
  }
  __syncthreads();

  float a = 0.f;
  for (int t = 0; t < T_; ++t)
    a += p[t] * Hm[((long)b * T_ + t) * (2 * H_) + tid];
  attended[b * (2 * H_) + tid] = a;
}

// ===========================================================================
// Group-routed decoders: 256 -> 64 -> 32 -> 1 for all 4 experts, pick label.
// ===========================================================================
__global__ void __launch_bounds__(256)
decoder(const float* __restrict__ attended, const int* __restrict__ labels,
        const float* __restrict__ w1, const float* __restrict__ b1,
        const float* __restrict__ w2, const float* __restrict__ b2,
        const float* __restrict__ w3, const float* __restrict__ b3,
        float* __restrict__ preds) {
  const int b = blockIdx.x, tid = threadIdx.x;
  __shared__ float av[2 * H_];
  __shared__ float h1s[4][64];
  __shared__ float h2s[4][32];
  __shared__ float outs[4];

  av[tid] = attended[b * (2 * H_) + tid];
  __syncthreads();
  {
    int e = tid >> 6, u = tid & 63;
    const float* w = w1 + (long)(e * 64 + u) * (2 * H_);
    float s = 0.f;
    for (int d = 0; d < 2 * H_; ++d) s += w[d] * av[d];
    s += b1[e * 64 + u];
    h1s[e][u] = s > 0.f ? s : 0.f;
  }
  __syncthreads();
  if (tid < 128) {
    int e = tid >> 5, v = tid & 31;
    const float* w = w2 + (long)(e * 32 + v) * 64;
    float s = 0.f;
    for (int u = 0; u < 64; ++u) s += w[u] * h1s[e][u];
    s += b2[e * 32 + v];
    h2s[e][v] = s > 0.f ? s : 0.f;
  }
  __syncthreads();
  if (tid < 4) {
    const float* w = w3 + tid * 32;
    float s = 0.f;
    for (int v = 0; v < 32; ++v) s += w[v] * h2s[tid][v];
    outs[tid] = s + b3[tid];
  }
  __syncthreads();
  if (tid == 0) preds[b] = outs[labels[b]];
}

// ===========================================================================
// Launch
// ===========================================================================
extern "C" void kernel_launch(void* const* d_in, const int* in_sizes, int n_in,
                              void* d_out, int out_size, void* d_ws, size_t ws_size,
                              hipStream_t stream) {
  (void)in_sizes; (void)n_in; (void)out_size; (void)ws_size;

  const float* x       = (const float*)d_in[0];
  const int*   labels  = (const int*)  d_in[1];
  const float* conv1_w = (const float*)d_in[2];
  const float* conv1_b = (const float*)d_in[3];
  const float* bn1_g   = (const float*)d_in[4];
  const float* bn1_b   = (const float*)d_in[5];
  const float* conv2_w = (const float*)d_in[6];
  const float* conv2_b = (const float*)d_in[7];
  const float* bn2_g   = (const float*)d_in[8];
  const float* bn2_b   = (const float*)d_in[9];
  const float* wih0    = (const float*)d_in[10];
  const float* whh0    = (const float*)d_in[11];
  const float* b0      = (const float*)d_in[12];
  const float* wih1    = (const float*)d_in[13];
  const float* whh1    = (const float*)d_in[14];
  const float* b1      = (const float*)d_in[15];
  const float* att_w   = (const float*)d_in[16];
  const float* att_b   = (const float*)d_in[17];
  const float* dec_w1  = (const float*)d_in[18];
  const float* dec_b1  = (const float*)d_in[19];
  const float* dec_w2  = (const float*)d_in[20];
  const float* dec_b2  = (const float*)d_in[21];
  const float* dec_w3  = (const float*)d_in[22];
  const float* dec_b3  = (const float*)d_in[23];

  // workspace layout (all regions 256B aligned)
  char* p = (char*)d_ws;
  auto alloc = [&](size_t bytes) -> void* {
    void* r = (void*)p;
    p += (bytes + 255) & ~(size_t)255;
    return r;
  };
  unsigned short* xb    = (unsigned short*)alloc((size_t)BT_ * 128 * 2);
  unsigned short* wt1   = (unsigned short*)alloc((size_t)3 * 64 * 128 * 2);
  unsigned short* wt2   = (unsigned short*)alloc((size_t)3 * 128 * 64 * 2);
  unsigned short* wih0b = (unsigned short*)alloc((size_t)2 * G_ * 128 * 2);
  unsigned short* whh0b = (unsigned short*)alloc((size_t)2 * G_ * H_ * 2);
  unsigned short* wih1b = (unsigned short*)alloc((size_t)2 * G_ * 256 * 2);
  unsigned short* whh1b = (unsigned short*)alloc((size_t)2 * G_ * H_ * 2);
  float*          y1f   = (float*)         alloc((size_t)BT_ * 64 * 4);
  unsigned short* y1b   = (unsigned short*)alloc((size_t)BT_ * 64 * 2);
  float*          y2f   = (float*)         alloc((size_t)BT_ * 128 * 4);
  unsigned short* y2b   = (unsigned short*)alloc((size_t)BT_ * 128 * 2);
  unsigned short* hb0   = (unsigned short*)alloc((size_t)BT_ * 256 * 2);
  float*          hf1   = (float*)         alloc((size_t)BT_ * 256 * 4);
  unsigned short* xgb   = (unsigned short*)alloc((size_t)2 * BT_ * G_ * 2);
  float*          stats = (float*)         alloc(384 * 4);
  float*          scores   = (float*)      alloc((size_t)BT_ * 4);
  float*          attended = (float*)      alloc((size_t)B_ * 256 * 4);

  float* preds = (float*)d_out;            // (B,1)
  float* attn  = (float*)d_out + B_;       // (B,T)

  // ---- bf16 pre-conversions ----
  f32_to_bf16<<<BT_ * 128 / 1024, 256, 0, stream>>>(x, xb);
  f32_to_bf16<<<2 * G_ * 128 / 1024, 256, 0, stream>>>(wih0, wih0b);
  f32_to_bf16<<<2 * G_ * H_ / 1024, 256, 0, stream>>>(whh0, whh0b);
  f32_to_bf16<<<2 * G_ * 256 / 1024, 256, 0, stream>>>(wih1, wih1b);
  f32_to_bf16<<<2 * G_ * H_ / 1024, 256, 0, stream>>>(whh1, whh1b);
  convw_tr_bf16<<<(64 * 128 * 3 + 255) / 256, 256, 0, stream>>>(conv1_w, wt1, 64, 128);
  convw_tr_bf16<<<(128 * 64 * 3 + 255) / 256, 256, 0, stream>>>(conv2_w, wt2, 128, 64);

  // ---- encoder: conv + BN + ReLU (x2) ----
  conv3_wmma<128, 64><<<BT_ / 16, 32, 0, stream>>>(xb, wt1, conv1_b, y1f);
  bn_stats<<<64, 256, 0, stream>>>(y1f, 64, BT_, stats + 0, stats + 64);
  bn_relu_bf16<<<BT_ * 64 / 256, 256, 0, stream>>>(y1f, 64, stats + 0, stats + 64,
                                                   bn1_g, bn1_b, y1b);
  conv3_wmma<64, 128><<<BT_ / 16, 32, 0, stream>>>(y1b, wt2, conv2_b, y2f);
  bn_stats<<<128, 256, 0, stream>>>(y2f, 128, BT_, stats + 128, stats + 256);
  bn_relu_bf16<<<BT_ * 128 / 256, 256, 0, stream>>>(y2f, 128, stats + 128, stats + 256,
                                                    bn2_g, bn2_b, y2b);

  // ---- BiLSTM layer 0 ----
  dim3 gg(BT_ / 16, G_ / 64);
  gate_gemm<<<gg, 32, 0, stream>>>(y2b, 128, wih0b,            b0,      xgb);
  gate_gemm<<<gg, 32, 0, stream>>>(y2b, 128, wih0b + G_ * 128, b0 + G_, xgb + (size_t)BT_ * G_);
  lstm_scan<<<16, 256, 0, stream>>>(xgb, whh0b, nullptr, hb0);

  // ---- BiLSTM layer 1 (input 2H = 256) ----
  gate_gemm<<<gg, 32, 0, stream>>>(hb0, 256, wih1b,            b1,      xgb);
  gate_gemm<<<gg, 32, 0, stream>>>(hb0, 256, wih1b + G_ * 256, b1 + G_, xgb + (size_t)BT_ * G_);
  lstm_scan<<<16, 256, 0, stream>>>(xgb, whh1b, hf1, nullptr);

  // ---- attention + decoders ----
  att_scores<<<BT_ / 256, 256, 0, stream>>>(hf1, att_w, att_b, scores);
  softmax_attend<<<B_, 256, 0, stream>>>(scores, hf1, attn, attended);
  decoder<<<B_, 256, 0, stream>>>(attended, labels, dec_w1, dec_b1, dec_w2, dec_b2,
                                  dec_w3, dec_b3, preds);
}
